// SelfAttention_32023276159712
// MI455X (gfx1250) — matrix operand order repaired and checked by reading, not compiled, not run on hardware
//
#include <hip/hip_runtime.h>

// ---------------- CDNA5 types ----------------
typedef __attribute__((ext_vector_type(16))) __bf16 v16bf;
typedef __attribute__((ext_vector_type(8)))  __bf16 v8bf;
typedef __attribute__((ext_vector_type(8)))  float  v8f;
typedef __attribute__((ext_vector_type(4)))  float  v4f;

#define NB   16      // batch
#define CIN  256
#define CA   32
#define CH   128
#define NSP  4096    // 64*64
#define NPL  1024    // 32*32

__device__ __forceinline__ v16bf cat8(v8bf lo, v8bf hi) {
    v16bf r;
#pragma unroll
    for (int i = 0; i < 8; ++i) { r[i] = lo[i]; r[i + 8] = hi[i]; }
    return r;
}

__device__ __forceinline__ v8f wmma_bf16(v16bf a, v16bf b, v8f c) {
    return __builtin_amdgcn_wmma_f32_16x16x32_bf16(false, a, false, b,
                                                   (short)0, c, false, false);
}

// ---------------- K0: weight prep (f32 -> bf16, pack) ----------------
// wall[192][256]: rows 0-31 = w_f1, 32-63 = w_f2, 64-191 = w_c1
__global__ void k_prep(const float* __restrict__ wf1, const float* __restrict__ wf2,
                       const float* __restrict__ wc1, const float* __restrict__ wc2,
                       __bf16* __restrict__ wall, __bf16* __restrict__ wc2b) {
    int i = blockIdx.x * blockDim.x + threadIdx.x;
    const int total1 = 192 * 256;
    if (i < total1) {
        int r = i >> 8, c = i & 255;
        float v;
        if (r < 32)      v = wf1[r * 256 + c];
        else if (r < 64) v = wf2[(r - 32) * 256 + c];
        else             v = wc1[(r - 64) * 256 + c];
        wall[i] = (__bf16)v;
    }
    int j = i - total1;
    if (j >= 0 && j < 256 * 128) wc2b[j] = (__bf16)wc2[j];
}

// ---------------- K1: projections via WMMA + shfl maxpool ----------------
// Each wave owns one 4x4 spatial tile; 16 columns = 4x4 block so the 2x2
// maxpool is two shfl_xor-max steps on the D tile.
__global__ void __launch_bounds__(128) k_proj(const float* __restrict__ x,
        const __bf16* __restrict__ wall,
        __bf16* __restrict__ f1T,    // [B][4096][32]
        __bf16* __restrict__ f2pT,   // [B][1024][32]
        float*  __restrict__ vp)     // [B][1024][128]
{
    const int lane = threadIdx.x & 31;
    const int wid  = threadIdx.x >> 5;
    const int tid  = blockIdx.x * 4 + wid;       // 0 .. 16*256-1
    const int b    = tid >> 8;
    const int t    = tid & 255;
    const int h0   = (t >> 4) << 2;
    const int w0   = (t & 15) << 2;
    const int nl   = lane & 15;
    const int half = lane >> 4;
    const int py   = nl >> 2, px = nl & 3;
    const int n    = (h0 + py) * 64 + (w0 + px);

    const float* xb = x + (size_t)b * CIN * NSP;

    v8f acc[12] = {};

#pragma unroll
    for (int kc = 0; kc < 8; ++kc) {
        // B operand: channels c = kc*32 + 16*half + e at spatial column n
        v16bf bx;
#pragma unroll
        for (int e = 0; e < 16; ++e) {
            int c = kc * 32 + half * 16 + e;
            bx[e] = (__bf16)xb[(size_t)c * NSP + n];
        }
#pragma unroll
        for (int m = 0; m < 12; ++m) {
            const __bf16* wr = wall + (m * 16 + nl) * 256 + kc * 32 + half * 16;
            v16bf av = cat8(*(const v8bf*)wr, *(const v8bf*)(wr + 8));
            acc[m] = wmma_bf16(av, bx, acc[m]);
        }
    }

    // f1 (m-tiles 0,1): store [b][n][ch], ch contiguous (B-operand friendly)
    {
        __bf16* f1r = f1T + ((size_t)b * NSP + n) * CA;
#pragma unroll
        for (int m = 0; m < 2; ++m) {
            v8bf o;
#pragma unroll
            for (int v = 0; v < 8; ++v) o[v] = (__bf16)acc[m][v];
            *(v8bf*)(f1r + m * 16 + half * 8) = o;
        }
    }

    const bool isPool = ((nl & 5) == 0);                      // px,py even
    const int  kp = (h0 / 2 + (py >> 1)) * 32 + (w0 / 2 + (px >> 1));

    // f2 pooled (m-tiles 2,3) -> [b][kp][ch] bf16
#pragma unroll
    for (int m = 2; m < 4; ++m) {
        v8bf o;
#pragma unroll
        for (int v = 0; v < 8; ++v) {
            float t0 = acc[m][v];
            t0 = fmaxf(t0, __shfl_xor(t0, 1, 32));   // pool along px
            t0 = fmaxf(t0, __shfl_xor(t0, 4, 32));   // pool along py
            o[v] = (__bf16)t0;
        }
        if (isPool)
            *(v8bf*)(f2pT + ((size_t)b * NPL + kp) * CA + (m - 2) * 16 + half * 8) = o;
    }

    // v pooled (m-tiles 4..11) -> [b][kp][ch] f32
#pragma unroll
    for (int m = 4; m < 12; ++m) {
        v4f o0, o1;
#pragma unroll
        for (int v = 0; v < 8; ++v) {
            float t0 = acc[m][v];
            t0 = fmaxf(t0, __shfl_xor(t0, 1, 32));
            t0 = fmaxf(t0, __shfl_xor(t0, 4, 32));
            if (v < 4) o0[v] = t0; else o1[v - 4] = t0;
        }
        if (isPool) {
            float* dst = vp + ((size_t)b * NPL + kp) * CH + (m - 4) * 16 + half * 8;
            *(v4f*)dst       = o0;
            *(v4f*)(dst + 4) = o1;
        }
    }
}

// ---------------- K2: softmax row stats + vt = v * (1/Z) ----------------
// One wave per (batch, 16 k-rows). Two WMMA passes over all 256 n-tiles:
// pass1 row max, pass2 exp-sum. Folds 1/Z into the value matrix (bf16).
__global__ void __launch_bounds__(128) k_stats(
        const __bf16* __restrict__ f1T,
        const __bf16* __restrict__ f2pT,
        const float*  __restrict__ vp,
        float*  __restrict__ mxg,    // [B][1024]
        __bf16* __restrict__ vt)     // [B][128][1024]
{
    __shared__ float rzl[4][16];
    const int lane = threadIdx.x & 31;
    const int wid  = threadIdx.x >> 5;
    const int gw   = blockIdx.x * 4 + wid;       // 0..1023
    const int b    = gw >> 6;
    const int kb   = (gw & 63) * 16;
    const int nl   = lane & 15, half = lane >> 4;

    const __bf16* ar = f2pT + ((size_t)b * NPL + kb + nl) * CA + half * 16;
    const v16bf as = cat8(*(const v8bf*)ar, *(const v8bf*)(ar + 8));
    const __bf16* f1b = f1T + (size_t)b * NSP * CA;
    const v8f zc = {};

    float mx[8];
#pragma unroll
    for (int v = 0; v < 8; ++v) mx[v] = -3.0e38f;

    for (int nt = 0; nt < 256; ++nt) {
        const __bf16* br = f1b + (size_t)(nt * 16 + nl) * CA + half * 16;
        v16bf bs = cat8(*(const v8bf*)br, *(const v8bf*)(br + 8));
        v8f s = wmma_bf16(as, bs, zc);
#pragma unroll
        for (int v = 0; v < 8; ++v) mx[v] = fmaxf(mx[v], s[v]);
    }
#pragma unroll
    for (int v = 0; v < 8; ++v) {
        mx[v] = fmaxf(mx[v], __shfl_xor(mx[v], 1, 32));
        mx[v] = fmaxf(mx[v], __shfl_xor(mx[v], 2, 32));
        mx[v] = fmaxf(mx[v], __shfl_xor(mx[v], 4, 32));
        mx[v] = fmaxf(mx[v], __shfl_xor(mx[v], 8, 32));
    }
    float Z[8] = {};
    for (int nt = 0; nt < 256; ++nt) {
        const __bf16* br = f1b + (size_t)(nt * 16 + nl) * CA + half * 16;
        v16bf bs = cat8(*(const v8bf*)br, *(const v8bf*)(br + 8));
        v8f s = wmma_bf16(as, bs, zc);
#pragma unroll
        for (int v = 0; v < 8; ++v) Z[v] += __expf(s[v] - mx[v]);
    }
#pragma unroll
    for (int v = 0; v < 8; ++v) {
        Z[v] += __shfl_xor(Z[v], 1, 32);
        Z[v] += __shfl_xor(Z[v], 2, 32);
        Z[v] += __shfl_xor(Z[v], 4, 32);
        Z[v] += __shfl_xor(Z[v], 8, 32);
    }
    if (nl == 0) {
#pragma unroll
        for (int v = 0; v < 8; ++v) {
            int k = v + 8 * half;                 // D row -> k
            mxg[(size_t)b * NPL + kb + k] = mx[v];
            rzl[wid][k] = 1.0f / Z[v];
        }
    }
    asm volatile("s_wait_dscnt 0" ::: "memory");  // intra-wave LDS RAW

    for (int i = lane; i < CH * 16; i += 32) {
        int h = i >> 4, kl = i & 15;
        vt[((size_t)b * CH + h) * NPL + kb + kl] =
            (__bf16)(vp[((size_t)b * NPL + kb + kl) * CH + h] * rzl[wid][kl]);
    }
}

// ---------------- K3: fused flash apply: O1 = vt * exp(s - mx) ----------------
// Wave owns 16 n columns x all 128 h rows; per 32-k chunk: 2 score wmma,
// exp, LDS D->B layout shuffle, 8 accumulation wmma.
__global__ void __launch_bounds__(128) k_apply(
        const __bf16* __restrict__ f1T,
        const __bf16* __restrict__ f2pT,
        const __bf16* __restrict__ vt,
        const float*  __restrict__ mxg,
        __bf16* __restrict__ O1T)    // [B][4096][128]
{
    __shared__ __align__(16) __bf16 lsP[4][16][32];
    const int lane = threadIdx.x & 31;
    const int wid  = threadIdx.x >> 5;
    const int gw   = blockIdx.x * 4 + wid;       // 0..4095
    const int b    = gw >> 8;
    const int n0   = (gw & 255) * 16;
    const int nl   = lane & 15, half = lane >> 4;

    const __bf16* br = f1T + ((size_t)b * NSP + n0 + nl) * CA + half * 16;
    const v16bf bs = cat8(*(const v8bf*)br, *(const v8bf*)(br + 8));
    const v8f zc = {};

    v8f acc[8] = {};

    const __bf16* f2b = f2pT + (size_t)b * NPL * CA;
    const float*  mxb = mxg  + (size_t)b * NPL;
    const __bf16* vtb = vt   + (size_t)b * CH * NPL;

    for (int kb = 0; kb < NPL; kb += 32) {
        // recompute 32x16 score tile as two 16x16 wmma, exp, stage in LDS
#pragma unroll
        for (int t = 0; t < 2; ++t) {
            const __bf16* arr = f2b + (size_t)(kb + t * 16 + nl) * CA + half * 16;
            v16bf as = cat8(*(const v8bf*)arr, *(const v8bf*)(arr + 8));
            v8f s = wmma_bf16(as, bs, zc);
            const float* mr = mxb + kb + t * 16 + half * 8;
            v4f m0 = *(const v4f*)mr;
            v4f m1 = *(const v4f*)(mr + 4);
            v8bf p;
#pragma unroll
            for (int v = 0; v < 8; ++v) {
                float m = (v < 4) ? m0[v] : m1[v - 4];
                p[v] = (__bf16)__expf(s[v] - m);
            }
            *(v8bf*)(&lsP[wid][nl][t * 16 + half * 8]) = p;   // [n][k] layout
        }
        asm volatile("s_wait_dscnt 0" ::: "memory");
        v16bf bp;
        {
            const __bf16* lp = &lsP[wid][nl][half * 16];
            bp = cat8(*(const v8bf*)lp, *(const v8bf*)(lp + 8));
        }
#pragma unroll
        for (int ht = 0; ht < 8; ++ht) {
            const __bf16* vr = vtb + (size_t)(ht * 16 + nl) * NPL + kb + half * 16;
            v16bf av = cat8(*(const v8bf*)vr, *(const v8bf*)(vr + 8));
            acc[ht] = wmma_bf16(av, bp, acc[ht]);
        }
    }

    __bf16* orow = O1T + ((size_t)b * NSP + n0 + nl) * CH;
#pragma unroll
    for (int ht = 0; ht < 8; ++ht) {
        v8bf o;
#pragma unroll
        for (int v = 0; v < 8; ++v) o[v] = (__bf16)acc[ht][v];
        *(v8bf*)(orow + ht * 16 + half * 8) = o;
    }
}

// ---------------- K4: y = x + sigma*(w_c2 * O1 + b) ----------------
__global__ void __launch_bounds__(128) k_final(
        const float*  __restrict__ x,
        const __bf16* __restrict__ O1T,
        const __bf16* __restrict__ wc2b,   // [256][128]
        const float*  __restrict__ b2,
        const float*  __restrict__ sigma,
        float* __restrict__ y)
{
    const int lane = threadIdx.x & 31;
    const int wid  = threadIdx.x >> 5;
    const int gw   = blockIdx.x * 4 + wid;       // 0..4095
    const int b    = gw >> 8;
    const int n0   = (gw & 255) * 16;
    const int nl   = lane & 15, half = lane >> 4;
    const float sg = sigma[0];

    v16bf bo[4];
    const __bf16* orow = O1T + ((size_t)b * NSP + n0 + nl) * CH;
#pragma unroll
    for (int kc = 0; kc < 4; ++kc) {
        const __bf16* p = orow + kc * 32 + half * 16;
        bo[kc] = cat8(*(const v8bf*)p, *(const v8bf*)(p + 8));
    }

    const float* xb = x + (size_t)b * CIN * NSP + n0 + nl;
    float*       yb = y + (size_t)b * CIN * NSP + n0 + nl;

#pragma unroll
    for (int mt = 0; mt < 16; ++mt) {
        v8f acc = {};
#pragma unroll
        for (int kc = 0; kc < 4; ++kc) {
            const __bf16* wr = wc2b + (mt * 16 + nl) * CH + kc * 32 + half * 16;
            v16bf aw = cat8(*(const v8bf*)wr, *(const v8bf*)(wr + 8));
            acc = wmma_bf16(aw, bo[kc], acc);
        }
        const float* br2 = b2 + mt * 16 + half * 8;
        v4f bb0 = *(const v4f*)br2;
        v4f bb1 = *(const v4f*)(br2 + 4);
#pragma unroll
        for (int v = 0; v < 8; ++v) {
            int d = mt * 16 + half * 8 + v;       // D row -> out channel
            float bias = (v < 4) ? bb0[v] : bb1[v - 4];
            size_t idx = (size_t)d * NSP;
            yb[idx] = xb[idx] + sg * (acc[v] + bias);
        }
    }
}

// ---------------- host ----------------
extern "C" void kernel_launch(void* const* d_in, const int* in_sizes, int n_in,
                              void* d_out, int out_size, void* d_ws, size_t ws_size,
                              hipStream_t stream) {
    const float* x   = (const float*)d_in[0];
    const float* wf1 = (const float*)d_in[1];
    const float* wf2 = (const float*)d_in[2];
    const float* wc1 = (const float*)d_in[3];
    const float* wc2 = (const float*)d_in[4];
    const float* b2  = (const float*)d_in[5];
    const float* sg  = (const float*)d_in[6];
    float* y = (float*)d_out;

    char* ws = (char*)d_ws;
    size_t off = 0;
    auto alloc = [&](size_t bytes) {
        void* p = ws + off;
        off = (off + bytes + 255) & ~(size_t)255;
        return p;
    };
    __bf16* wall = (__bf16*)alloc((size_t)192 * 256 * 2);
    __bf16* wc2b = (__bf16*)alloc((size_t)256 * 128 * 2);
    __bf16* f1T  = (__bf16*)alloc((size_t)NB * NSP * CA * 2);
    __bf16* f2pT = (__bf16*)alloc((size_t)NB * NPL * CA * 2);
    float*  vp   = (float*) alloc((size_t)NB * NPL * CH * 4);
    float*  mxg  = (float*) alloc((size_t)NB * NPL * 4);
    __bf16* vt   = (__bf16*)alloc((size_t)NB * CH * NPL * 2);
    __bf16* O1T  = (__bf16*)alloc((size_t)NB * NSP * CH * 2);

    k_prep <<<320, 256, 0, stream>>>(wf1, wf2, wc1, wc2, wall, wc2b);
    k_proj <<<1024, 128, 0, stream>>>(x, wall, f1T, f2pT, vp);
    k_stats<<<256, 128, 0, stream>>>(f1T, f2pT, vp, mxg, vt);
    k_apply<<<1024, 128, 0, stream>>>(f1T, f2pT, vt, mxg, O1T);
    k_final<<<1024, 128, 0, stream>>>(x, O1T, wc2b, b2, sg, y);
}
